// MultilayerGRU_44306882625757
// MI455X (gfx1250) — matrix operand order repaired and checked by reading, not compiled, not run on hardware
//
#include <hip/hip_runtime.h>
#include <hip/hip_bf16.h>
#include <cstdint>

// ---------------------------------------------------------------------------
// Types for CDNA5 WMMA (wave32)
// ---------------------------------------------------------------------------
typedef __attribute__((ext_vector_type(16))) __bf16 v16bf;
typedef __attribute__((ext_vector_type(8)))  __bf16 v8bf;
typedef __attribute__((ext_vector_type(8)))  float  v8f;

#define B_   64
#define S_   512
#define I_   512
#define H_   1024
#define O_   512
#define NH3  3072   // 3*H (z | r | g concatenated along N)
#define NWG_SCAN 96

// ---------------------------------------------------------------------------
// Helpers
// ---------------------------------------------------------------------------
__device__ __forceinline__ unsigned short f2bf(float f) {
  unsigned u = __builtin_bit_cast(unsigned, f);
  u += 0x7FFFu + ((u >> 16) & 1u);          // round-to-nearest-even
  return (unsigned short)(u >> 16);
}
__device__ __forceinline__ float sigmoidf_(float x) {
  return 1.0f / (1.0f + __expf(-x));
}

// A fragment (16x32 bf16, row-major source).  Layout: lanes 0-15 hold
// K=[0..15]; lanes 16-31 hold K=[16..31] (matches contiguous B fragment).
__device__ __forceinline__ v16bf load_a_frag(const unsigned short* Arow, int k0, int kAoff) {
  v8bf alo = *(const v8bf*)(Arow + k0 + kAoff);        // 16B
  v8bf ahi = *(const v8bf*)(Arow + k0 + kAoff + 8);    // 16B
  return __builtin_shufflevector(alo, ahi, 0,1,2,3,4,5,6,7,8,9,10,11,12,13,14,15);
}

__device__ __forceinline__ v8f wmma_bf16(v16bf a, v16bf b, v8f c) {
  // v_wmma_f32_16x16x32_bf16 (neg_a, A, neg_b, B, c_mod, C, reuse_a, reuse_b)
  return __builtin_amdgcn_wmma_f32_16x16x32_bf16(false, a, false, b, (short)0, c, false, false);
}

// gfx1250 async global->LDS copy (16B per lane, tracked by ASYNCcnt).
// The VDST operand is a 32-bit LDS byte address; low 32 bits of the generic
// address of a __shared__ object are exactly that offset (flat aperture).
__device__ __forceinline__ void async_copy_b128(unsigned short* lds_dst,
                                                const unsigned short* gsrc) {
  unsigned ldsa = (unsigned)(uintptr_t)lds_dst;
  asm volatile("global_load_async_to_lds_b128 %0, %1, off"
               :: "v"(ldsa), "v"(gsrc)
               : "memory");
}
__device__ __forceinline__ void wait_async0() {
  asm volatile("s_wait_asynccnt 0x0" ::: "memory");
}

// Device-wide arrive/spin barrier for the persistent scan kernel.
__device__ __forceinline__ void gsync(unsigned* cnt, unsigned target) {
  __threadfence();
  __syncthreads();
  if (threadIdx.x == 0) {
    atomicAdd(cnt, 1u);
    while (atomicAdd(cnt, 0u) < target) __builtin_amdgcn_s_sleep(2);
  }
  __syncthreads();
  __threadfence();
}

// ---------------------------------------------------------------------------
// Elementwise fp32 -> bf16 convert (row-major, for A matrices)
// ---------------------------------------------------------------------------
__global__ void k_cvt_bf16(unsigned short* __restrict__ dst,
                           const float* __restrict__ src, int n) {
  int i = blockIdx.x * blockDim.x + threadIdx.x;
  if (i < n) dst[i] = f2bf(src[i]);
}

// Pack fp32 [K,N] row-major weight -> bf16 column-major [N cols x K]
// (so each B fragment is a contiguous 32B load per lane).
__global__ void k_pack_cm(unsigned short* __restrict__ dst,
                          const float* __restrict__ src, int K, int N) {
  int i = blockIdx.x * blockDim.x + threadIdx.x;
  if (i >= K * N) return;
  int n = i / K;
  int k = i - n * K;
  dst[i] = f2bf(src[(size_t)k * N + n]);
}

// Init h / h_bf16 from h0[:, layer, :]; also zeroes the scan sync counter.
__global__ void k_init_h(float* __restrict__ h, unsigned short* __restrict__ hbf,
                         const float* __restrict__ h0, int layer,
                         unsigned* __restrict__ cnt) {
  int i = blockIdx.x * blockDim.x + threadIdx.x;
  if (i == 0) *cnt = 0u;
  if (i < B_ * H_) {
    int b = i / H_;
    int k = i - b * H_;
    float v = h0[((size_t)b * 2 + layer) * H_ + k];
    h[i] = v;
    hbf[i] = f2bf(v);
  }
}

// ---------------------------------------------------------------------------
// Batched WMMA GEMM: C[M,N](fp32,ldc) = A[M,K](bf16,rm) * B(bf16,cm) + bias[N]
// Each wave computes a 64x32 block (4 M-tiles x 2 N-tiles, 8 accumulators):
// per k-step 4 A-fragments + 2 B-fragments feed 8 WMMAs.
// ---------------------------------------------------------------------------
__global__ void __launch_bounds__(256) k_gemm_bf16(
    float* __restrict__ C, int ldc,
    const unsigned short* __restrict__ A, int lda,
    const unsigned short* __restrict__ Bcm,
    const float* __restrict__ bias,
    int M, int N, int K) {
  const int wave = threadIdx.x >> 5;
  const int lane = threadIdx.x & 31;
  const int blocksN = N >> 5;                       // 32-wide N blocks
  const int blocksM = M >> 6;                       // 64-tall M blocks
  const int wb = blockIdx.x * 8 + wave;
  if (wb >= blocksM * blocksN) return;

  const int bm = (wb / blocksN) << 6;
  const int bn = (wb - (wb / blocksN) * blocksN) << 5;

  const int kAoff = (lane >> 4) << 4;   // 0 or 16
  const int kBoff = (lane >> 4) << 4;   // 0 or 16
  const unsigned short* Arow[4];
#pragma unroll
  for (int i = 0; i < 4; ++i)
    Arow[i] = A + (size_t)(bm + i * 16 + (lane & 15)) * lda;
  const unsigned short* Bcol[2];
#pragma unroll
  for (int j = 0; j < 2; ++j)
    Bcol[j] = Bcm + (size_t)(bn + j * 16 + (lane & 15)) * K;

  v8f acc[4][2] = {};
  for (int k0 = 0; k0 < K; k0 += 32) {
    __builtin_prefetch(Arow[0] + k0 + 256, 0, 1);   // global_prefetch_b8
    v16bf b0 = *(const v16bf*)(Bcol[0] + k0 + kBoff);
    v16bf b1 = *(const v16bf*)(Bcol[1] + k0 + kBoff);
#pragma unroll
    for (int i = 0; i < 4; ++i) {
      v16bf a = load_a_frag(Arow[i], k0, kAoff);
      acc[i][0] = wmma_bf16(a, b0, acc[i][0]);
      acc[i][1] = wmma_bf16(a, b1, acc[i][1]);
    }
  }

  const int mHi = (lane >> 4) << 3;
#pragma unroll
  for (int j = 0; j < 2; ++j) {
    const int nCol = bn + j * 16 + (lane & 15);
    const float bv = bias ? bias[nCol] : 0.0f;
#pragma unroll
    for (int i = 0; i < 4; ++i)
#pragma unroll
      for (int v = 0; v < 8; ++v)
        C[(size_t)(bm + i * 16 + v + mHi) * ldc + nCol] = acc[i][j][v] + bv;
  }
}

// ---------------------------------------------------------------------------
// Persistent GRU scan over S timesteps for one layer.
// Grid = 96 WGs x 64 threads (2 waves).  WG w owns 32 columns of
// [W_hz|W_hr|W_hg] (bf16, column-major) resident in 64KB LDS (staged with
// global_load_async_to_lds_b128).  Each wave owns 16 columns and computes all
// 4 M-tiles, so each B fragment (LDS) feeds 4 WMMAs.
// Phase A (WGs 0..63): z, r, rh.   Phase B (WGs 64..95): g, h update.
// ---------------------------------------------------------------------------
__global__ void __launch_bounds__(64) k_gru_scan(
    const float* __restrict__ proj,            // [B*S, 3072] fp32 (xz|xr|xg)
    const unsigned short* __restrict__ whcat,  // [3072 cols x 1024] bf16 cm
    float* __restrict__ h,                     // [B,H] fp32
    unsigned short* __restrict__ hbf,          // [B,H] bf16
    unsigned short* __restrict__ rh,           // [B,H] bf16 (r*h)
    float* __restrict__ zbuf,                  // [B,H] fp32
    unsigned short* __restrict__ seq,          // [B*S, H] bf16 layer output
    float* __restrict__ hid_out,               // hidden out base (+layer*H)
    unsigned* __restrict__ cnt) {
  __shared__ unsigned short wlds[32 * H_];     // 64 KB of resident weights

  const int wg      = blockIdx.x;              // 0..95
  const int colBase = wg << 5;                 // first of this WG's 32 columns

  // Async-stage this WG's weight slice into LDS (contiguous, column-major).
  {
    const unsigned short* src = whcat + (size_t)colBase * H_;
    for (int c = threadIdx.x; c < (32 * H_) / 8; c += 64)
      async_copy_b128(&wlds[c * 8], src + c * 8);  // 16B per lane
    wait_async0();
  }
  __syncthreads();

  const int wave  = threadIdx.x >> 5;          // 0..1
  const int lane  = threadIdx.x & 31;
  const int nloc  = wave << 4;                 // 0 or 16 within WG's 32 cols
  const int jcol  = colBase + nloc + (lane & 15);
  const int kAoff = (lane >> 4) << 4;
  const int kBoff = (lane >> 4) << 4;
  const int mHi   = (lane >> 4) << 3;
  const unsigned short* Bcol = wlds + (size_t)(nloc + (lane & 15)) * H_;
  const bool isA = (colBase < 2 * H_);

  unsigned epoch = 0;
  for (int t = 0; t < S_; ++t) {
    if (isA) {
      v8f acc[4];
#pragma unroll
      for (int mt = 0; mt < 4; ++mt)
#pragma unroll
        for (int v = 0; v < 8; ++v)
          acc[mt][v] = proj[((size_t)(mt * 16 + v + mHi) * S_ + t) * NH3 + jcol];
#pragma unroll 2
      for (int k0 = 0; k0 < H_; k0 += 32) {
        v16bf b = *(const v16bf*)(Bcol + k0 + kBoff);
#pragma unroll
        for (int mt = 0; mt < 4; ++mt) {
          v16bf a = load_a_frag(hbf + (size_t)(mt * 16 + (lane & 15)) * H_, k0, kAoff);
          acc[mt] = wmma_bf16(a, b, acc[mt]);
        }
      }
      if (jcol < H_) {                         // z gate
#pragma unroll
        for (int mt = 0; mt < 4; ++mt)
#pragma unroll
          for (int v = 0; v < 8; ++v) {
            int m = mt * 16 + v + mHi;
            zbuf[(size_t)m * H_ + jcol] = sigmoidf_(acc[mt][v]);
          }
      } else {                                 // r gate -> rh = bf16(r*h)
        const int hcol = jcol - H_;
#pragma unroll
        for (int mt = 0; mt < 4; ++mt)
#pragma unroll
          for (int v = 0; v < 8; ++v) {
            int m = mt * 16 + v + mHi;
            float r = sigmoidf_(acc[mt][v]);
            rh[(size_t)m * H_ + hcol] = f2bf(r * h[(size_t)m * H_ + hcol]);
          }
      }
    }
    gsync(cnt, (++epoch) * NWG_SCAN);

    if (!isA) {                                // g gate + h update
      const int hcol = jcol - 2 * H_;
      v8f acc[4];
#pragma unroll
      for (int mt = 0; mt < 4; ++mt)
#pragma unroll
        for (int v = 0; v < 8; ++v)
          acc[mt][v] = proj[((size_t)(mt * 16 + v + mHi) * S_ + t) * NH3 + jcol];
#pragma unroll 2
      for (int k0 = 0; k0 < H_; k0 += 32) {
        v16bf b = *(const v16bf*)(Bcol + k0 + kBoff);
#pragma unroll
        for (int mt = 0; mt < 4; ++mt) {
          v16bf a = load_a_frag(rh + (size_t)(mt * 16 + (lane & 15)) * H_, k0, kAoff);
          acc[mt] = wmma_bf16(a, b, acc[mt]);
        }
      }
#pragma unroll
      for (int mt = 0; mt < 4; ++mt)
#pragma unroll
        for (int v = 0; v < 8; ++v) {
          int m = mt * 16 + v + mHi;
          float g  = tanhf(acc[mt][v]);
          float ho = h[(size_t)m * H_ + hcol];
          float z  = zbuf[(size_t)m * H_ + hcol];
          float hn = z * ho + (1.0f - z) * g;
          h[(size_t)m * H_ + hcol]   = hn;
          hbf[(size_t)m * H_ + hcol] = f2bf(hn);
          seq[((size_t)m * S_ + t) * H_ + hcol] = f2bf(hn);
          if (t == S_ - 1) hid_out[(size_t)m * 2 * H_ + hcol] = hn;
        }
    }
    gsync(cnt, (++epoch) * NWG_SCAN);
  }
}

// ---------------------------------------------------------------------------
// Host-side orchestration
// ---------------------------------------------------------------------------
extern "C" void kernel_launch(void* const* d_in, const int* in_sizes, int n_in,
                              void* d_out, int out_size, void* d_ws, size_t ws_size,
                              hipStream_t stream) {
  (void)in_sizes; (void)n_in; (void)out_size; (void)ws_size;

  const float* x   = (const float*)d_in[0];
  const float* h0  = (const float*)d_in[1];
  const float* Wxz0 = (const float*)d_in[2];
  const float* Whz0 = (const float*)d_in[3];
  const float* bz0  = (const float*)d_in[4];
  const float* Wxr0 = (const float*)d_in[5];
  const float* Whr0 = (const float*)d_in[6];
  const float* br0  = (const float*)d_in[7];
  const float* Wxg0 = (const float*)d_in[8];
  const float* Whg0 = (const float*)d_in[9];
  const float* bg0  = (const float*)d_in[10];
  const float* Wxz1 = (const float*)d_in[11];
  const float* Whz1 = (const float*)d_in[12];
  const float* bz1  = (const float*)d_in[13];
  const float* Wxr1 = (const float*)d_in[14];
  const float* Whr1 = (const float*)d_in[15];
  const float* br1  = (const float*)d_in[16];
  const float* Wxg1 = (const float*)d_in[17];
  const float* Whg1 = (const float*)d_in[18];
  const float* bg1  = (const float*)d_in[19];
  const float* Why  = (const float*)d_in[20];
  const float* by   = (const float*)d_in[21];

  float* out_seq = (float*)d_out;                              // [B,S,O]
  float* out_hid = out_seq + (size_t)B_ * S_ * O_;             // [B,L,H]

  char* ws = (char*)d_ws;
  size_t off = 0;
  auto alloc = [&](size_t bytes) -> void* {
    off = (off + 255) & ~(size_t)255;
    void* p = ws + off;
    off += bytes;
    return p;
  };
  unsigned*       cnt  = (unsigned*)alloc(sizeof(unsigned));
  float*          proj = (float*)alloc((size_t)B_ * S_ * NH3 * 4);
  unsigned short* xbf  = (unsigned short*)alloc((size_t)B_ * S_ * I_ * 2);
  unsigned short* seq0 = (unsigned short*)alloc((size_t)B_ * S_ * H_ * 2);
  unsigned short* seq1 = (unsigned short*)alloc((size_t)B_ * S_ * H_ * 2);
  unsigned short* wx0  = (unsigned short*)alloc((size_t)I_ * NH3 * 2);
  unsigned short* wx1  = (unsigned short*)alloc((size_t)H_ * NH3 * 2);
  unsigned short* wh0  = (unsigned short*)alloc((size_t)H_ * NH3 * 2);
  unsigned short* wh1  = (unsigned short*)alloc((size_t)H_ * NH3 * 2);
  unsigned short* why  = (unsigned short*)alloc((size_t)H_ * O_ * 2);
  float*          hbuf = (float*)alloc((size_t)B_ * H_ * 4);
  unsigned short* hbf  = (unsigned short*)alloc((size_t)B_ * H_ * 2);
  unsigned short* rh   = (unsigned short*)alloc((size_t)B_ * H_ * 2);
  float*          zbuf = (float*)alloc((size_t)B_ * H_ * 4);

  auto pack = [&](unsigned short* dst, const float* src, int K, int N) {
    int n = K * N;
    k_pack_cm<<<(n + 255) / 256, 256, 0, stream>>>(dst, src, K, N);
  };
  auto gemm = [&](float* C, int ldc, const unsigned short* A, int lda,
                  const unsigned short* Bcm, const float* bias, int M, int N, int K) {
    int blocks = ((M >> 6) * (N >> 5) + 7) / 8;
    k_gemm_bf16<<<blocks, 256, 0, stream>>>(C, ldc, A, lda, Bcm, bias, M, N, K);
  };

  // ---- 1. Convert x to bf16; pack all weights bf16 column-major ----
  {
    int n = B_ * S_ * I_;
    k_cvt_bf16<<<(n + 255) / 256, 256, 0, stream>>>(xbf, x, n);
  }
  pack(wx0 + 0 * (size_t)H_ * I_, Wxz0, I_, H_);
  pack(wx0 + 1 * (size_t)H_ * I_, Wxr0, I_, H_);
  pack(wx0 + 2 * (size_t)H_ * I_, Wxg0, I_, H_);
  pack(wh0 + 0 * (size_t)H_ * H_, Whz0, H_, H_);
  pack(wh0 + 1 * (size_t)H_ * H_, Whr0, H_, H_);
  pack(wh0 + 2 * (size_t)H_ * H_, Whg0, H_, H_);
  pack(wx1 + 0 * (size_t)H_ * H_, Wxz1, H_, H_);
  pack(wx1 + 1 * (size_t)H_ * H_, Wxr1, H_, H_);
  pack(wx1 + 2 * (size_t)H_ * H_, Wxg1, H_, H_);
  pack(wh1 + 0 * (size_t)H_ * H_, Whz1, H_, H_);
  pack(wh1 + 1 * (size_t)H_ * H_, Whr1, H_, H_);
  pack(wh1 + 2 * (size_t)H_ * H_, Whg1, H_, H_);
  pack(why, Why, H_, O_);

  const int M = B_ * S_;                       // 32768
  // ---- 2. Layer 0 input projections ----
  gemm(proj + 0 * H_, NH3, xbf, I_, wx0 + 0 * (size_t)H_ * I_, bz0, M, H_, I_);
  gemm(proj + 1 * H_, NH3, xbf, I_, wx0 + 1 * (size_t)H_ * I_, br0, M, H_, I_);
  gemm(proj + 2 * H_, NH3, xbf, I_, wx0 + 2 * (size_t)H_ * I_, bg0, M, H_, I_);
  // ---- 3. Layer 0 scan ----
  k_init_h<<<(B_ * H_ + 255) / 256, 256, 0, stream>>>(hbuf, hbf, h0, 0, cnt);
  k_gru_scan<<<NWG_SCAN, 64, 0, stream>>>(proj, wh0, hbuf, hbf, rh, zbuf, seq0, out_hid + 0 * H_, cnt);
  // ---- 4. Layer 1 input projections ----
  gemm(proj + 0 * H_, NH3, seq0, H_, wx1 + 0 * (size_t)H_ * H_, bz1, M, H_, H_);
  gemm(proj + 1 * H_, NH3, seq0, H_, wx1 + 1 * (size_t)H_ * H_, br1, M, H_, H_);
  gemm(proj + 2 * H_, NH3, seq0, H_, wx1 + 2 * (size_t)H_ * H_, bg1, M, H_, H_);
  // ---- 5. Layer 1 scan ----
  k_init_h<<<(B_ * H_ + 255) / 256, 256, 0, stream>>>(hbuf, hbf, h0, 1, cnt);
  k_gru_scan<<<NWG_SCAN, 64, 0, stream>>>(proj, wh1, hbuf, hbf, rh, zbuf, seq1, out_hid + 1 * H_, cnt);
  // ---- 6. Output projection ----
  gemm(out_seq, O_, seq1, H_, why, by, M, O_, H_);
}
